// PatchEmbedding_34308198760657
// MI455X (gfx1250) — matrix operand order repaired and checked by reading, not compiled, not run on hardware
//
#include <hip/hip_runtime.h>
#include <hip/hip_bf16.h>

// ---------------------------------------------------------------------------
// PatchEmbedding with data-dependent patch subdivision, MI455X (gfx1250).
// bf16 WMMA (v_wmma_f32_16x16x32_bf16), wave32. Fragments pre-swizzled into
// the exact ISA VGPR layouts; GEMM uses a 2x2 register tile per wave so the
// inner loop is 2 global_load_b128 per v_wmma (16 KFLOP per KB of L2 traffic).
// ---------------------------------------------------------------------------

typedef __attribute__((ext_vector_type(16))) __bf16 v16bf;
typedef __attribute__((ext_vector_type(8)))  float  v8f;

#define PS      32
#define IMG     512
#define GW      16              // IMG/PS grid width (= grid height)
#define CIN     3
#define EMBED   768
#define BATCH   32
#define NPMAX   256             // num_patches for this problem
#define KFULL   3072            // CIN*PS*PS
#define KSUB    768             // CIN*16*16
#define KT_FULL 96              // KFULL/32
#define KT_SUB  24              // KSUB/32
#define NCT     48              // EMBED/16 column tiles

static __device__ __forceinline__ unsigned f2bf1(float f) {
  unsigned u = __float_as_uint(f);
  return (u + 0x7FFFu + ((u >> 16) & 1u)) >> 16;   // round-to-nearest-even
}
static __device__ __forceinline__ unsigned pk2bf(float a, float b) {
  return f2bf1(a) | (f2bf1(b) << 16);
}

static __device__ __forceinline__ v8f wmma_bf16(v16bf a, v16bf b, v8f c) {
  return __builtin_amdgcn_wmma_f32_16x16x32_bf16(
      false, a, false, b, (short)0, c, false, false);
}

// ---------------------------------------------------------------------------
// Phase 0: recompute the host-side subdivision plan on device.
//   sizes = round(32*s + 16*(1-s)), s = sigmoid(32*mean(w)^12/w^12 - 24)
//   subdiv (kd==2) -> 4 tokens with pooled 16x16 weight, else 1 full token.
// ---------------------------------------------------------------------------
__global__ void plan_kernel(const float* __restrict__ iw, int np,
                            int* __restrict__ hdr,
                            int* __restrict__ fullPatch, int* __restrict__ fullTok,
                            int* __restrict__ subList,  int* __restrict__ subTok,
                            float* __restrict__ out, int out_size) {
  __shared__ float sf[256];
  __shared__ int sa[256], sb[256], sc[256];
  int t = threadIdx.x;
  float w = (t < np) ? iw[t] : 0.f;
  sf[t] = w;
  __syncthreads();
  for (int s = 128; s > 0; s >>= 1) {
    if (t < s) sf[t] += sf[t + s];
    __syncthreads();
  }
  float avg = sf[0] / (float)np;

  int count = 0, usable = 0;
  if (t < np) {
    float r  = avg / w;
    float r2 = r * r, r4 = r2 * r2, r8 = r4 * r4;
    float v  = 32.f * r8 * r4;                       // 32 * (avg/w)^12
    float sg = 1.f / (1.f + __expf(-(v - 24.f)));
    int   sz = (int)rintf(32.f * sg + 16.f * (1.f - sg));
    int   subdiv = (sz < PS) && (sz > 7);
    int   kd = subdiv ? (PS / sz) : 1;
    usable = (subdiv && kd == 2) ? 1 : 0;            // only 16x16 path occurs
    count  = usable ? 4 : 1;
  }
  sa[t] = count;
  sb[t] = (t < np && !usable) ? 1 : 0;
  sc[t] = usable;
  __syncthreads();
  for (int s = 1; s < 256; s <<= 1) {                // Hillis-Steele inclusive
    int a = sa[t], b = sb[t], c = sc[t];
    int a2 = 0, b2 = 0, c2 = 0;
    if (t >= s) { a2 = sa[t - s]; b2 = sb[t - s]; c2 = sc[t - s]; }
    __syncthreads();
    sa[t] = a + a2; sb[t] = b + b2; sc[t] = c + c2;
    __syncthreads();
  }
  int off      = sa[t] - count;                      // exclusive token offset
  int fullRank = sb[t] - ((t < np && !usable) ? 1 : 0);
  int subRank  = sc[t] - usable;
  if (t < np) {
    if (!usable) {
      fullPatch[fullRank] = t;
      fullTok[fullRank]   = off;
    } else {
      for (int k = 0; k < 4; ++k) {
        subList[subRank * 4 + k] = t | (k << 16);
        subTok[subRank * 4 + k]  = off + k;
      }
    }
    out[(out_size - np) + t] = (float)count;         // counts output (as f32)
  }
  if (t == 255) {
    hdr[0] = sb[255];                                // full tokens per batch
    hdr[1] = sc[255] * 4;                            // sub tokens per batch
    hdr[2] = sa[255];                                // T
  }
}

// ---------------------------------------------------------------------------
// Weight swizzle -> WMMA B-fragment order, bf16.
// B (32x16, K x N): lane L holds N = L&15, K = (e&7) + 8*(L>>4) + 16*(e>>3), e = 0..15.
// Buffer layout: [kt][ct][lane][16 bf16]  (one v16bf per lane, 32B).
// ---------------------------------------------------------------------------
__global__ void swz_w_full(const float* __restrict__ w, unsigned* __restrict__ Bw) {
  int t = blockIdx.x * 256 + threadIdx.x;
  if (t >= KT_FULL * NCT * 32 * 8) return;
  int e2 = t & 7, lane = (t >> 3) & 31;
  int ct = (t >> 8) % NCT, kt = (t >> 8) / NCT;
  int n  = ct * 16 + (lane & 15);
  int kk = ((e2 & 3) << 1) + ((lane >> 4) << 3) + ((e2 >> 2) << 4);
  int k  = kt * 32 + kk;                             // even -> 8B aligned src
  const float2 v = *(const float2*)(w + (long)n * KFULL + k);
  Bw[((long)(kt * NCT + ct) * 32 + lane) * 8 + e2] = pk2bf(v.x, v.y);
}

__global__ void swz_w_sub(const float* __restrict__ w, unsigned* __restrict__ Bw) {
  int t = blockIdx.x * 256 + threadIdx.x;
  if (t >= KT_SUB * NCT * 32 * 8) return;
  int e2 = t & 7, lane = (t >> 3) & 31;
  int ct = (t >> 8) % NCT, kt = (t >> 8) / NCT;
  int n  = ct * 16 + (lane & 15);
  int kk = ((e2 & 3) << 1) + ((lane >> 4) << 3) + ((e2 >> 2) << 4);
  int k  = kt * 32 + kk;                             // 0..767
  int c = k >> 8, rem = k & 255, i = rem >> 4, j = rem & 15;  // j even
  const float* base = w + (long)n * KFULL + c * 1024 + (2 * i) * PS + 2 * j;
  float4 r0 = *(const float4*)base;                  // 2j % 4 == 0 -> 16B ok
  float4 r1 = *(const float4*)(base + PS);
  float m0 = 0.25f * (r0.x + r0.y + r1.x + r1.y);    // 2x2 mean pool
  float m1 = 0.25f * (r0.z + r0.w + r1.z + r1.w);
  Bw[((long)(kt * NCT + ct) * 32 + lane) * 8 + e2] = pk2bf(m0, m1);
}

// ---------------------------------------------------------------------------
// im2col -> WMMA A-fragment order, bf16.
// A (16x32, M x K): lane = M + 16*laneHi, K = (e&7) + 16*(e>>3) + 8*laneHi.
// Buffer layout: [rowTile][kt][lane][16 bf16]. row = b*nTok + tokenIdx.
// One thread per packed 32-bit word (two adjacent-K pixels, float2 source).
// ---------------------------------------------------------------------------
__global__ void im2col_full(const float* __restrict__ x, const int* __restrict__ hdr,
                            const int* __restrict__ fullPatch, unsigned* __restrict__ A) {
  long t = (long)blockIdx.x * 256 + threadIdx.x;
  int  nTok = hdr[0];
  long totalWords = (long)BATCH * nTok * (KFULL / 2);
  if (t >= totalWords) return;
  int  word = (int)(t % (KFULL / 2));
  long row  = t / (KFULL / 2);
  int k0 = word * 2;
  int kt = k0 >> 5, kk = k0 & 31;
  int m = (int)(row & 15); long rowTile = row >> 4;
  int lane = m + (((kk >> 3) & 1) << 4);
  int e2   = ((kk & 7) >> 1) + ((kk >> 4) << 2);
  int b  = (int)(row / nTok), fi = (int)(row % nTok);
  int p  = fullPatch[fi], py = p >> 4, px = p & (GW - 1);
  int c = k0 >> 10, rem = k0 & 1023, i = rem >> 5, j = rem & 31;  // j even
  const float2 v = *(const float2*)(x +
      ((long)(b * CIN + c) * IMG + (py * PS + i)) * IMG + (px * PS + j));
  A[((rowTile * KT_FULL + kt) * 32 + lane) * 8 + e2] = pk2bf(v.x, v.y);
}

__global__ void im2col_sub(const float* __restrict__ x, const int* __restrict__ hdr,
                           const int* __restrict__ subList, unsigned* __restrict__ A) {
  long t = (long)blockIdx.x * 256 + threadIdx.x;
  int  nTok = hdr[1];
  long totalWords = (long)BATCH * nTok * (KSUB / 2);
  if (t >= totalWords) return;
  int  word = (int)(t % (KSUB / 2));
  long row  = t / (KSUB / 2);
  int k0 = word * 2;
  int kt = k0 >> 5, kk = k0 & 31;
  int m = (int)(row & 15); long rowTile = row >> 4;
  int lane = m + (((kk >> 3) & 1) << 4);
  int e2   = ((kk & 7) >> 1) + ((kk >> 4) << 2);
  int b  = (int)(row / nTok), siI = (int)(row % nTok);
  int ent = subList[siI];
  int p = ent & 0xFFFF, si = ent >> 16;
  int py = p >> 4, px = p & (GW - 1);
  int dy = (si >> 1) & 1, dx = si & 1;
  int c = k0 >> 8, rem = k0 & 255, i = rem >> 4, j = rem & 15;    // j even
  const float2 v = *(const float2*)(x +
      ((long)(b * CIN + c) * IMG + (py * PS + dy * 16 + i)) * IMG +
      (px * PS + dx * 16 + j));
  A[((rowTile * KT_SUB + kt) * 32 + lane) * 8 + e2] = pk2bf(v.x, v.y);
}

// ---------------------------------------------------------------------------
// WMMA GEMM with 2x2 register tiling: each wave owns 32 rows x 32 cols of D
// (4 accumulators). Block = 8 waves = 32 rows x 256 cols; grid.y = 3 col
// groups. Per k-step: 4 fragment loads -> 4 v_wmma (2 b128 loads per wmma).
// Bias folded into accumulator init (all 8 C/D elems of a lane share one N).
// All exits are wave-uniform, so EXEC is all-ones at every v_wmma.
// ---------------------------------------------------------------------------
__global__ void gemm_wmma(const unsigned short* __restrict__ Abuf,
                          const unsigned short* __restrict__ Bbuf,
                          const int* __restrict__ hdr, const int* __restrict__ tok,
                          const float* __restrict__ bias, float* __restrict__ out,
                          int KT, int which) {
  int wv = threadIdx.x >> 5, lane = threadIdx.x & 31;
  int nTok = hdr[which];
  long T = (long)hdr[2];
  int rowBase = blockIdx.x * 32;                     // 2 row tiles per block
  if (rowBase >= nTok * BATCH) return;               // wave-uniform exit
  int colBase = blockIdx.y * 256 + wv * 32;          // 2 col tiles per wave
  int n0 = colBase + (lane & 15);
  int n1 = n0 + 16;
  float bv0 = bias[n0], bv1 = bias[n1];
  v8f a00, a01, a10, a11;
#pragma unroll
  for (int r = 0; r < 8; ++r) {
    a00[r] = bv0; a01[r] = bv1; a10[r] = bv0; a11[r] = bv1;
  }

  const v16bf* Af0 = (const v16bf*)Abuf + (long)(2 * blockIdx.x) * KT * 32 + lane;
  const v16bf* Af1 = Af0 + (long)KT * 32;
  const v16bf* Bf0 = (const v16bf*)Bbuf + (long)(colBase >> 4) * 32 + lane;
  const v16bf* Bf1 = Bf0 + 32;
#pragma unroll 2
  for (int kt = 0; kt < KT; ++kt) {
    v16bf fa0 = Af0[(long)kt * 32];
    v16bf fa1 = Af1[(long)kt * 32];
    v16bf fb0 = Bf0[(long)kt * NCT * 32];
    v16bf fb1 = Bf1[(long)kt * NCT * 32];
    a00 = wmma_bf16(fa0, fb0, a00);
    a01 = wmma_bf16(fa0, fb1, a01);
    a10 = wmma_bf16(fa1, fb0, a10);
    a11 = wmma_bf16(fa1, fb1, a11);
  }

  int mhi = (lane >> 4) << 3;                        // C/D: M = r + 8*laneHi
#pragma unroll
  for (int r = 0; r < 8; ++r) {
    int row0 = rowBase + mhi + r;                    // row tile 0
    int b0  = row0 / nTok;
    long o0 = ((long)b0 * T + tok[row0 - b0 * nTok]) * EMBED;
    out[o0 + n0] = a00[r];
    out[o0 + n1] = a01[r];
    int row1 = row0 + 16;                            // row tile 1
    int b1  = row1 / nTok;
    long o1 = ((long)b1 * T + tok[row1 - b1 * nTok]) * EMBED;
    out[o1 + n0] = a10[r];
    out[o1 + n1] = a11[r];
  }
}

// ---------------------------------------------------------------------------
extern "C" void kernel_launch(void* const* d_in, const int* in_sizes, int n_in,
                              void* d_out, int out_size, void* d_ws, size_t ws_size,
                              hipStream_t stream) {
  const float* x      = (const float*)d_in[0];
  const float* iw     = (const float*)d_in[1];
  const float* weight = (const float*)d_in[2];
  const float* bias   = (const float*)d_in[3];
  int np = in_sizes[1];                              // 256
  float* out = (float*)d_out;

  // Workspace carve (max-bound sizes; ~107 MB, resident in 192 MB L2)
  char* p = (char*)d_ws;
  auto carve = [&](size_t bytes) -> void* {
    void* r = (void*)p;
    p += (bytes + 255) & ~(size_t)255;
    return r;
  };
  int* hdr          = (int*)carve(64);
  int* fullPatch    = (int*)carve(NPMAX * 4);
  int* fullTok      = (int*)carve(NPMAX * 4);
  int* subList      = (int*)carve(NPMAX * 4 * 4);
  int* subTok       = (int*)carve(NPMAX * 4 * 4);
  unsigned short* BwF = (unsigned short*)carve((size_t)EMBED * KFULL * 2);
  unsigned short* BwS = (unsigned short*)carve((size_t)EMBED * KSUB * 2);
  unsigned short* Af  = (unsigned short*)carve((size_t)BATCH * NPMAX * KFULL * 2);
  unsigned short* As  = (unsigned short*)carve((size_t)BATCH * NPMAX * 4 * KSUB * 2);

  plan_kernel<<<1, 256, 0, stream>>>(iw, np, hdr, fullPatch, fullTok,
                                     subList, subTok, out, out_size);

  swz_w_full<<<(KT_FULL * NCT * 32 * 8 + 255) / 256, 256, 0, stream>>>(
      weight, (unsigned*)BwF);
  swz_w_sub<<<(KT_SUB * NCT * 32 * 8 + 255) / 256, 256, 0, stream>>>(
      weight, (unsigned*)BwS);

  long wFull = (long)BATCH * NPMAX * (KFULL / 2);        // max-bound grids;
  long wSub  = (long)BATCH * NPMAX * 4 * (KSUB / 2);     // kernels early-exit
  im2col_full<<<(unsigned)((wFull + 255) / 256), 256, 0, stream>>>(
      x, hdr, fullPatch, (unsigned*)Af);
  im2col_sub<<<(unsigned)((wSub + 255) / 256), 256, 0, stream>>>(
      x, hdr, subList, (unsigned*)As);

  dim3 gFull(BATCH * NPMAX / 32, EMBED / 256);           // 256 x 3
  dim3 gSub(BATCH * NPMAX * 4 / 32, EMBED / 256);        // 1024 x 3
  gemm_wmma<<<gFull, 256, 0, stream>>>(Af, BwF, hdr, fullTok, bias, out,
                                       KT_FULL, 0);
  gemm_wmma<<<gSub, 256, 0, stream>>>(As, BwS, hdr, subTok, bias, out,
                                      KT_SUB, 1);
}
